// TTT_78451872629262
// MI455X (gfx1250) — matrix operand order repaired and checked by reading, not compiled, not run on hardware
//
#include <hip/hip_runtime.h>
#include <math.h>

#define DEVINL __device__ __forceinline__

typedef __attribute__((ext_vector_type(16))) __bf16 v16bf;
typedef __attribute__((ext_vector_type(8)))  float  v8f;

union ABFrag { v16bf v; uint4 q[2]; unsigned short u[16]; };
union U4U16  { uint4 q; unsigned short u[8]; };

DEVINL unsigned short f2bf(float f) {
  unsigned int u = __float_as_uint(f);
  u += 0x7FFFu + ((u >> 16) & 1u);      // round-to-nearest-even
  return (unsigned short)(u >> 16);
}

DEVINL v8f wmma_bf16(const ABFrag& a, const ABFrag& b, v8f c) {
  // (neg_a, A, neg_b, B, c_mod, C, reuse_a, reuse_b)
  return __builtin_amdgcn_wmma_f32_16x16x32_bf16(false, a.v, false, b.v,
                                                 (short)0, c, false, false);
}

// Problem constants
constexpr int B_  = 32, N_ = 576, C_ = 768, NH_ = 12, D_ = 64;
constexpr int QKVN = 3 * C_ + 3 * D_;   // 2496
constexpr int XCN  = C_ + D_;           // 832
constexpr float ESC = -(1.0f / 3.0f) / 576.0f;  // -SCALE / n, SCALE = 9^-0.5

// ---------------------------------------------------------------------------
// fp32 -> bf16 conversion (grid-stride)
// ---------------------------------------------------------------------------
__global__ __launch_bounds__(256) void k_f2bf(const float* __restrict__ s,
                                              unsigned short* __restrict__ d,
                                              long n) {
  long i  = (long)blockIdx.x * blockDim.x + threadIdx.x;
  long st = (long)gridDim.x * blockDim.x;
  for (; i < n; i += st) d[i] = f2bf(s[i]);
}

// ---------------------------------------------------------------------------
// C[M,N] = A[M,K](bf16) @ B[K,N](bf16) + bias[N] ; C fp32
// block: 256 threads (8 waves); block tile 256(M) x 64(N); K step 32.
// Wave wv owns rows [wv*32, wv*32+32): 2 A-frags x 4 B-frags = 8 WMMAs/step.
// Global->LDS staging is software-pipelined through registers so LOADcnt
// latency overlaps the WMMA burst.
// ---------------------------------------------------------------------------
__global__ __launch_bounds__(256) void k_gemm(const unsigned short* __restrict__ A,
                                              const unsigned short* __restrict__ Bm,
                                              const float* __restrict__ bias,
                                              float* __restrict__ Cm,
                                              int M, int Nn, int K) {
  __shared__ unsigned short sA[256][40];   // [m][k] row-major, padded
  __shared__ unsigned short sB[64][40];    // [n][k] transposed, padded
  const int tid = threadIdx.x;
  const int l = tid & 31, wv = tid >> 5;
  const int row0 = blockIdx.y * 256, n0 = blockIdx.x * 64;
  (void)M;

  v8f acc[2][4] = {};
  // Staging assignments: A -> one 32-elem row per thread; B -> 8-elem strip.
  const int br = tid >> 3, bc = (tid & 7) * 8;
  const int koff = (l < 16) ? 0 : 8;
  const int kb   = (l < 16) ? 0 : 16;
  const int nIn  = l & 15;
  const int arow = wv * 32 + nIn;

  // Prefetch first K-tile into registers.
  uint4 pa0, pa1, pa2, pa3, pb;
  {
    const uint4* ga = (const uint4*)(A + (size_t)(row0 + tid) * K);
    pa0 = ga[0]; pa1 = ga[1]; pa2 = ga[2]; pa3 = ga[3];
    pb = *(const uint4*)(Bm + (size_t)br * Nn + n0 + bc);
  }

  for (int kt = 0; kt < K; kt += 32) {
    __syncthreads();                       // previous compute done with LDS
    // Publish current tile to LDS.
    uint4* da = (uint4*)&sA[tid][0];
    da[0] = pa0; da[1] = pa1; da[2] = pa2; da[3] = pa3;
    U4U16 t; t.q = pb;
#pragma unroll
    for (int i = 0; i < 8; ++i) sB[bc + i][br] = t.u[i];
    __syncthreads();

    // Issue next tile's global loads (overlaps with WMMA below).
    if (kt + 32 < K) {
      const uint4* ga = (const uint4*)(A + (size_t)(row0 + tid) * K + kt + 32);
      pa0 = ga[0]; pa1 = ga[1]; pa2 = ga[2]; pa3 = ga[3];
      pb = *(const uint4*)(Bm + (size_t)(kt + 32 + br) * Nn + n0 + bc);
    }

    ABFrag a0, a1;
    a0.q[0] = *(const uint4*)&sA[arow][koff];
    a0.q[1] = *(const uint4*)&sA[arow][koff + 16];
    a1.q[0] = *(const uint4*)&sA[arow + 16][koff];
    a1.q[1] = *(const uint4*)&sA[arow + 16][koff + 16];
#pragma unroll
    for (int tn = 0; tn < 4; ++tn) {
      ABFrag bb;
      const int e = tn * 16 + nIn;
      bb.q[0] = *(const uint4*)&sB[e][koff];
      bb.q[1] = *(const uint4*)&sB[e][koff + 16];
      acc[0][tn] = wmma_bf16(a0, bb, acc[0][tn]);
      acc[1][tn] = wmma_bf16(a1, bb, acc[1][tn]);
    }
  }
#pragma unroll
  for (int sm = 0; sm < 2; ++sm) {
#pragma unroll
    for (int tn = 0; tn < 4; ++tn) {
      const int ng = n0 + tn * 16 + nIn;
      const float bv = bias[ng];
#pragma unroll
      for (int r = 0; r < 8; ++r) {
        const int mg = row0 + wv * 32 + sm * 16 + (l >> 4) * 8 + r;
        Cm[(size_t)mg * Nn + ng] = acc[sm][tn][r] + bv;
      }
    }
  }
}

// ---------------------------------------------------------------------------
// Per-(batch,head) SwiGLU fast-weight update + apply.
// 384 blocks, 256 threads (8 waves). Streams n in 9 chunks of 64.
// ---------------------------------------------------------------------------
__global__ __launch_bounds__(256) void k_swiglu(const float* __restrict__ qkv,
                                                const float* __restrict__ w1g,
                                                const float* __restrict__ w2g,
                                                unsigned short* __restrict__ xcat) {
  __shared__ unsigned short sW1T[64][72];  // W1^T  [e][d]
  __shared__ unsigned short sW2T[64][72];  // W2^T  [e][d]
  __shared__ unsigned short sK [64][72];   // K / Q chunk [n][d]
  __shared__ unsigned short sKT[64][72];   // K^T chunk   [d][n]
  __shared__ unsigned short sT1[64][72];   // T1^T [e][n], later W1u^T [e][d]
  __shared__ unsigned short sT2[64][72];   // T2^T [e][n], later W2u^T [e][d]
  __shared__ float sN1[64], sN2[64];

  const int tid = threadIdx.x;
  const int l = tid & 31, wv = tid >> 5;
  const int bh = blockIdx.x;
  const int b = bh / NH_, h = bh % NH_;
  const int q1off = h * 64, k1off = C_ + h * 64, v1off = 2 * C_ + h * 64;
  const float* w1h = w1g + h * 4096;
  const float* w2h = w2g + h * 4096;

  for (int idx = tid; idx < 4096; idx += 256) {
    const int d = idx >> 6, e = idx & 63;
    sW1T[e][d] = f2bf(w1h[idx]);
    sW2T[e][d] = f2bf(w2h[idx]);
  }

  v8f g1a[2] = {}, g2a[2] = {};
  const int koff = (l < 16) ? 0 : 8;
  const int kb   = (l < 16) ? 0 : 16;
  const int nIn  = l & 15, mb = (l >> 4) * 8;

  for (int ci = 0; ci < 9; ++ci) {
    const int n0 = ci * 64;
    __syncthreads();
    for (int idx = tid; idx < 4096; idx += 256) {
      const int r = idx >> 6, d = idx & 63;
      const unsigned short us =
          f2bf(qkv[(size_t)(b * N_ + n0 + r) * QKVN + k1off + d]);
      sK[r][d] = us; sKT[d][r] = us;
    }
    __syncthreads();
#pragma unroll
    for (int i = 0; i < 2; ++i) {
      const int t = wv * 2 + i, tm = t >> 2, tn = t & 3;
      v8f z1 = {}, z2 = {};
#pragma unroll
      for (int d0 = 0; d0 < 64; d0 += 32) {
        ABFrag a, b1, b2;
        const int row = tm * 16 + nIn;
        a.q[0] = *(const uint4*)&sK[row][d0 + koff];
        a.q[1] = *(const uint4*)&sK[row][d0 + koff + 16];
        const int e = tn * 16 + nIn;
        b1.q[0] = *(const uint4*)&sW1T[e][d0 + koff];
        b1.q[1] = *(const uint4*)&sW1T[e][d0 + koff + 16];
        b2.q[0] = *(const uint4*)&sW2T[e][d0 + koff];
        b2.q[1] = *(const uint4*)&sW2T[e][d0 + koff + 16];
        z1 = wmma_bf16(a, b1, z1);
        z2 = wmma_bf16(a, b2, z2);
      }
      const int e = tn * 16 + nIn;
#pragma unroll
      for (int r = 0; r < 8; ++r) {
        const int nr = tm * 16 + mb + r;
        const float vv  = qkv[(size_t)(b * N_ + n0 + nr) * QKVN + v1off + e];
        const float ec  = vv * ESC;
        const float z2v = z2[r];
        const float sg  = 1.0f / (1.0f + __expf(-z2v));
        const float t1  = ec * z2v * sg;                                   // e * silu(z2)
        const float t2  = ec * z1[r] * (sg * (1.0f + z2v * (1.0f - sg)));  // e*z1*silu'(z2)
        sT1[e][nr] = f2bf(t1);
        sT2[e][nr] = f2bf(t2);
      }
    }
    __syncthreads();
#pragma unroll
    for (int i = 0; i < 2; ++i) {
      const int t = wv * 2 + i, tm = t >> 2, tn = t & 3;
#pragma unroll
      for (int c0 = 0; c0 < 64; c0 += 32) {
        ABFrag a, b1, b2;
        const int drow = tm * 16 + nIn;
        a.q[0] = *(const uint4*)&sKT[drow][c0 + koff];
        a.q[1] = *(const uint4*)&sKT[drow][c0 + koff + 16];
        const int e = tn * 16 + nIn;
        b1.q[0] = *(const uint4*)&sT1[e][c0 + koff];
        b1.q[1] = *(const uint4*)&sT1[e][c0 + koff + 16];
        b2.q[0] = *(const uint4*)&sT2[e][c0 + koff];
        b2.q[1] = *(const uint4*)&sT2[e][c0 + koff + 16];
        g1a[i] = wmma_bf16(a, b1, g1a[i]);
        g2a[i] = wmma_bf16(a, b2, g2a[i]);
      }
    }
  }
  __syncthreads();
  if (tid < 64) { sN1[tid] = 0.f; sN2[tid] = 0.f; }
  __syncthreads();
#pragma unroll
  for (int i = 0; i < 2; ++i) {
    const int t = wv * 2 + i, tn = t & 3;
    const int e = tn * 16 + nIn;
    float p1 = 0.f, p2 = 0.f;
#pragma unroll
    for (int r = 0; r < 8; ++r) {
      p1 += g1a[i][r] * g1a[i][r];
      p2 += g2a[i][r] * g2a[i][r];
    }
    atomicAdd(&sN1[e], p1);
    atomicAdd(&sN2[e], p2);
  }
  __syncthreads();
#pragma unroll
  for (int i = 0; i < 2; ++i) {
    const int t = wv * 2 + i, tm = t >> 2, tn = t & 3;
    const int e = tn * 16 + nIn;
    const float f1 = 1.0f / (sqrtf(sN1[e]) + 1.0f);
    const float f2 = 1.0f / (sqrtf(sN2[e]) + 1.0f);
#pragma unroll
    for (int r = 0; r < 8; ++r) {
      const int d = tm * 16 + mb + r;
      sT1[e][d] = f2bf(w1h[d * 64 + e] - g1a[i][r] * f1);   // W1u^T
      sT2[e][d] = f2bf(w2h[d * 64 + e] - g2a[i][r] * f2);   // W2u^T
    }
  }
  // Apply: x1 = (Q @ W1u) * silu(Q @ W2u)
  for (int ci = 0; ci < 9; ++ci) {
    const int n0 = ci * 64;
    __syncthreads();
    for (int idx = tid; idx < 4096; idx += 256) {
      const int r = idx >> 6, d = idx & 63;
      sK[r][d] = f2bf(qkv[(size_t)(b * N_ + n0 + r) * QKVN + q1off + d]);
    }
    __syncthreads();
#pragma unroll
    for (int i = 0; i < 2; ++i) {
      const int t = wv * 2 + i, tm = t >> 2, tn = t & 3;
      v8f y1 = {}, y2 = {};
#pragma unroll
      for (int d0 = 0; d0 < 64; d0 += 32) {
        ABFrag a, b1, b2;
        const int row = tm * 16 + nIn;
        a.q[0] = *(const uint4*)&sK[row][d0 + koff];
        a.q[1] = *(const uint4*)&sK[row][d0 + koff + 16];
        const int e = tn * 16 + nIn;
        b1.q[0] = *(const uint4*)&sT1[e][d0 + koff];
        b1.q[1] = *(const uint4*)&sT1[e][d0 + koff + 16];
        b2.q[0] = *(const uint4*)&sT2[e][d0 + koff];
        b2.q[1] = *(const uint4*)&sT2[e][d0 + koff + 16];
        y1 = wmma_bf16(a, b1, y1);
        y2 = wmma_bf16(a, b2, y2);
      }
      const int e = tn * 16 + nIn;
#pragma unroll
      for (int r = 0; r < 8; ++r) {
        const int nr = tm * 16 + mb + r;
        const float yy = y2[r];
        const float s  = 1.0f / (1.0f + __expf(-yy));
        xcat[(size_t)(b * N_ + n0 + nr) * XCN + h * 64 + e] = f2bf(y1[r] * yy * s);
      }
    }
  }
}

// ---------------------------------------------------------------------------
// Depthwise 3x3 fast-weight update + conv apply. One wave per (b, channel).
// ---------------------------------------------------------------------------
__global__ __launch_bounds__(256) void k_dwc(const float* __restrict__ qkv,
                                             const float* __restrict__ w3g,
                                             unsigned short* __restrict__ xcat) {
  __shared__ float sKp[8][26][26];
  __shared__ float sQp[8][26][26];
  const int tid = threadIdx.x;
  const int l = tid & 31, wv = tid >> 5;
  const int p = blockIdx.x * 8 + wv;
  const int b = p >> 6, c = p & 63;
  const int q2off = 3 * C_, k2off = 3 * C_ + D_, v2off = 3 * C_ + 2 * D_;

  float* kf = &sKp[wv][0][0];
  float* qf = &sQp[wv][0][0];
  for (int i = l; i < 676; i += 32) { kf[i] = 0.f; qf[i] = 0.f; }
  __syncthreads();
  for (int i = l; i < 576; i += 32) {
    const int y = i / 24, x = i - y * 24;
    const size_t base = (size_t)(b * N_ + i) * QKVN;
    sKp[wv][y + 1][x + 1] = qkv[base + k2off + c];
    sQp[wv][y + 1][x + 1] = qkv[base + q2off + c];
  }
  __syncthreads();

  float g[9];
#pragma unroll
  for (int t = 0; t < 9; ++t) g[t] = 0.f;
  for (int i = l; i < 576; i += 32) {
    const int y = i / 24, x = i - y * 24;
    const float e = qkv[(size_t)(b * N_ + i) * QKVN + v2off + c] * ESC;
#pragma unroll
    for (int t = 0; t < 9; ++t) {
      const int dy = t / 3, dx = t - dy * 3;
      g[t] += sKp[wv][y + dy][x + dx] * e;
    }
  }
#pragma unroll
  for (int t = 0; t < 9; ++t) {
#pragma unroll
    for (int off = 16; off >= 1; off >>= 1) g[t] += __shfl_xor(g[t], off, 32);
  }
  float n2 = 0.f;
#pragma unroll
  for (int t = 0; t < 9; ++t) n2 += g[t] * g[t];
  const float fac = 1.0f / (sqrtf(n2) + 1.0f);
  float w3u[9];
#pragma unroll
  for (int t = 0; t < 9; ++t) w3u[t] = w3g[c * 9 + t] - g[t] * fac;

  for (int i = l; i < 576; i += 32) {
    const int y = i / 24, x = i - y * 24;
    float s = 0.f;
#pragma unroll
    for (int t = 0; t < 9; ++t) {
      const int dy = t / 3, dx = t - dy * 3;
      s += w3u[t] * sQp[wv][y + dy][x + dx];
    }
    xcat[(size_t)(b * N_ + i) * XCN + C_ + c] = f2bf(s);
  }
}

// ---------------------------------------------------------------------------
// Host orchestration
// ---------------------------------------------------------------------------
extern "C" void kernel_launch(void* const* d_in, const int* in_sizes, int n_in,
                              void* d_out, int out_size, void* d_ws, size_t ws_size,
                              hipStream_t stream) {
  (void)in_sizes; (void)n_in; (void)out_size; (void)ws_size;
  const float* x     = (const float*)d_in[0];
  // d_in[1]=h, d_in[2]=w (24, hardcoded)
  const float* qkvw  = (const float*)d_in[3];
  const float* qkvb  = (const float*)d_in[4];
  const float* w1    = (const float*)d_in[5];
  const float* w2    = (const float*)d_in[6];
  const float* w3    = (const float*)d_in[7];
  const float* projw = (const float*)d_in[8];
  const float* projb = (const float*)d_in[9];
  float* out = (float*)d_out;

  // Workspace layout (256-B aligned offsets)
  char* ws = (char*)d_ws;
  unsigned short* xb   = (unsigned short*)(ws);                 // x   bf16: 28,311,552 B
  unsigned short* qwb  = (unsigned short*)(ws + 28311552);      // qkv_w bf16: 3,833,856 B
  unsigned short* pwb  = (unsigned short*)(ws + 32145408);      // proj_w bf16: 1,277,952 B
  float*          qkv  = (float*)         (ws + 33423360);      // qkv fp32: 184,025,088 B
  unsigned short* xcat = (unsigned short*)(ws + 217448448);     // concat bf16: 30,670,848 B

  k_f2bf<<<2048, 256, 0, stream>>>(x,     xb,  (long)B_ * N_ * C_);
  k_f2bf<<<512,  256, 0, stream>>>(qkvw,  qwb, (long)C_ * QKVN);
  k_f2bf<<<256,  256, 0, stream>>>(projw, pwb, (long)XCN * C_);

  k_gemm<<<dim3(QKVN / 64, (B_ * N_) / 256), 256, 0, stream>>>(
      xb, qwb, qkvb, qkv, B_ * N_, QKVN, C_);

  k_swiglu<<<B_ * NH_, 256, 0, stream>>>(qkv, w1, w2, xcat);
  k_dwc<<<(B_ * D_) / 8, 256, 0, stream>>>(qkv, w3, xcat);

  k_gemm<<<dim3(C_ / 64, (B_ * N_) / 256), 256, 0, stream>>>(
      xcat, pwb, projb, out, B_ * N_, C_, XCN);
}
